// RegionLayer_1932735283345
// MI455X (gfx1250) — matrix operand order repaired and checked by reading, not compiled, not run on hardware
//
#include <hip/hip_runtime.h>

typedef __attribute__((ext_vector_type(16))) _Float16 v16h;
typedef __attribute__((ext_vector_type(8)))  float    v8f;

#define RL_EPS 1e-5f

// ------------------------------------------------------------------
// Kernel 1: per-(region, channel) stats -> scale/shift (deterministic)
// grid: 512 blocks (16 regions * 32 channels), 256 threads
// ------------------------------------------------------------------
__global__ __launch_bounds__(256) void rl_stats_kernel(
    const float* __restrict__ x, const float* __restrict__ gamma,
    const float* __restrict__ beta, float* __restrict__ stats /* [512][2] */)
{
  const int r  = blockIdx.x >> 5;   // region 0..15
  const int c  = blockIdx.x & 31;   // channel 0..31
  const int rr = r >> 2, rc = r & 3;
  const int tid = threadIdx.x;
  const int rowbase = (rr * 64) * 256 + rc * 64;

  float s = 0.f, sq = 0.f;
  for (int b = 0; b < 32; ++b) {
    const float* p = x + ((b * 32 + c) << 16) + rowbase;
    for (int t = tid; t < 4096; t += 256) {
      float v = p[(t >> 6) * 256 + (t & 63)];
      s += v; sq += v * v;
    }
  }

  __shared__ float s1[256], s2[256];
  s1[tid] = s; s2[tid] = sq;
  __syncthreads();
  for (int off = 128; off > 0; off >>= 1) {
    if (tid < off) { s1[tid] += s1[tid + off]; s2[tid] += s2[tid + off]; }
    __syncthreads();
  }
  if (tid == 0) {
    const float inv_n = 1.0f / 131072.0f;
    float mean  = s1[0] * inv_n;
    float var   = s2[0] * inv_n - mean * mean;
    float scale = gamma[c] * rsqrtf(var + RL_EPS);
    float shift = beta[c] - mean * scale;
    stats[(r * 32 + c) * 2 + 0] = scale;
    stats[(r * 32 + c) * 2 + 1] = shift;
  }
}

// ------------------------------------------------------------------
// Kernel 2: normalize + PReLU + 3x3 region conv via f16 WMMA implicit GEMM
// grid: 32 batches * 16 regions * 16 patches = 8192 blocks, 256 threads
// ------------------------------------------------------------------
__global__ __launch_bounds__(256) void rl_conv_kernel(
    const float* __restrict__ x, const float* __restrict__ stats,
    const float* __restrict__ prelu_a, const float* __restrict__ conv_w,
    const float* __restrict__ conv_b, float* __restrict__ out)
{
  const int bid = blockIdx.x;
  const int b  = bid >> 8;          // batch
  const int r  = (bid >> 4) & 15;   // region
  const int p  = bid & 15;          // 16x16 patch within region
  const int rr = r >> 2, rc = r & 3;
  const int py = p >> 2, px = p & 3;
  const int tid = threadIdx.x;

  // A-fragments: [cotile][tap][lane][16 f16]  = 2*9*32*16 = 9216 f16 (18 KB)
  __shared__ _Float16 Alds[9216];
  // Halo patch, channel-innermost: [ly 0..17][lx 0..17][ci 0..31] (20.25 KB)
  __shared__ _Float16 Xp[18 * 18 * 32];
  __shared__ float sc[32], sh[32];

  if (tid < 32) {
    sc[tid] = stats[(r * 32 + tid) * 2 + 0];
    sh[tid] = stats[(r * 32 + tid) * 2 + 1];
  }

  // Pre-swizzle weights into A-fragments, contiguous k order (k = 16*half + j, same as B).
  // f = ((ct*9 + tap)*32 + lane)*16 + j ; lane = M-row (mod 16) / K-half select
  for (int f = tid; f < 9216; f += 256) {
    int j    = f & 15;
    int lane = (f >> 4) & 31;
    int tap  = (f >> 9) % 9;
    int ct   = f / 4608;
    int m     = lane & 15;
    int kbase = (lane < 16) ? 0 : 16;
    int k     = kbase + j;   // k = input channel
    int co    = ct * 16 + m;
    Alds[f] = (_Float16)conv_w[(co * 32 + k) * 9 + tap];
  }

  const float a = prelu_a[0];
  __syncthreads();   // sc/sh visible

  // Fill halo patch: normalize + PReLU + f16 convert. Zero outside region.
  const int gy0 = rr * 64, gx0 = rc * 64;
  const int ry0 = py * 16 - 1, rx0 = px * 16 - 1;
  for (int ci = 0; ci < 32; ++ci) {
    const float* xc = x + ((b * 32 + ci) << 16);
    const float scale = sc[ci], shift = sh[ci];
    for (int t = tid; t < 324; t += 256) {
      int ly = t / 18, lx = t % 18;
      int ry = ry0 + ly, rx = rx0 + lx;
      float y = 0.f;
      if (ry >= 0 && ry < 64 && rx >= 0 && rx < 64) {
        float v = xc[(gy0 + ry) * 256 + (gx0 + rx)];
        y = v * scale + shift;
        y = (y > 0.f) ? y : a * y;
      }
      Xp[(ly * 18 + lx) * 32 + ci] = (_Float16)y;
    }
  }
  __syncthreads();

  // ---------------- WMMA phase (uniform control flow, EXEC all 1s) --------
  const int wave = tid >> 5, lane = tid & 31;
  const int n = lane & 15, half = lane >> 4;

  // Accumulator init = conv bias (C/D layout: m = i + 8*half, n = lane%16)
  v8f bias0, bias1;
#pragma unroll
  for (int i = 0; i < 8; ++i) {
    bias0[i] = conv_b[i + half * 8];
    bias1[i] = conv_b[16 + i + half * 8];
  }

  const v16h* Afrag = (const v16h*)Alds;
  const int pr0 = wave * 2, pr1 = wave * 2 + 1;   // two pixel rows per wave

  v8f acc00 = bias0, acc01 = bias1;   // row0: cotile0 / cotile1
  v8f acc10 = bias0, acc11 = bias1;   // row1

#pragma unroll
  for (int tap = 0; tap < 9; ++tap) {
    const int kh = tap / 3, kw = tap % 3;
    v16h A0 = Afrag[tap * 32 + lane];         // cotile 0
    v16h A1 = Afrag[288 + tap * 32 + lane];   // cotile 1
    // B fragment: 16 consecutive channels at (row, col) -> one 32B LDS read
    v16h B0 = *(const v16h*)(Xp + (((pr0 + kh) * 18 + (n + kw)) * 32 + half * 16));
    v16h B1 = *(const v16h*)(Xp + (((pr1 + kh) * 18 + (n + kw)) * 32 + half * 16));

    acc00 = __builtin_amdgcn_wmma_f32_16x16x32_f16(false, A0, false, B0,
                                                   (short)0, acc00, false, false);
    acc10 = __builtin_amdgcn_wmma_f32_16x16x32_f16(false, A0, false, B1,
                                                   (short)0, acc10, false, false);
    acc01 = __builtin_amdgcn_wmma_f32_16x16x32_f16(false, A1, false, B0,
                                                   (short)0, acc01, false, false);
    acc11 = __builtin_amdgcn_wmma_f32_16x16x32_f16(false, A1, false, B1,
                                                   (short)0, acc11, false, false);
  }

  // Store: out[b][co][gy][gx]; lane n -> consecutive columns (coalesced)
  const int gyb = gy0 + py * 16;
  const int gxb = gx0 + px * 16 + n;
#pragma unroll
  for (int i = 0; i < 8; ++i) {
    int co0 = i + half * 8;
    int co1 = 16 + i + half * 8;
    out[((b * 32 + co0) << 16) + (gyb + pr0) * 256 + gxb] = acc00[i];
    out[((b * 32 + co1) << 16) + (gyb + pr0) * 256 + gxb] = acc01[i];
    out[((b * 32 + co0) << 16) + (gyb + pr1) * 256 + gxb] = acc10[i];
    out[((b * 32 + co1) << 16) + (gyb + pr1) * 256 + gxb] = acc11[i];
  }
}

extern "C" void kernel_launch(void* const* d_in, const int* in_sizes, int n_in,
                              void* d_out, int out_size, void* d_ws, size_t ws_size,
                              hipStream_t stream) {
  (void)in_sizes; (void)n_in; (void)out_size; (void)ws_size;
  const float* x       = (const float*)d_in[0];
  const float* gamma   = (const float*)d_in[1];
  const float* beta    = (const float*)d_in[2];
  const float* prelu_a = (const float*)d_in[3];
  const float* conv_w  = (const float*)d_in[4];
  const float* conv_b  = (const float*)d_in[5];
  float* stats = (float*)d_ws;          // 1024 floats
  float* out   = (float*)d_out;

  rl_stats_kernel<<<512, 256, 0, stream>>>(x, gamma, beta, stats);
  rl_conv_kernel<<<8192, 256, 0, stream>>>(x, stats, prelu_a, conv_w, conv_b, out);
}
